// ScanLayer_65712999629423
// MI455X (gfx1250) — compile-verified, matrix operand order checked by reading, not run on hardware
//
#include <hip/hip_runtime.h>
#include <hip/hip_bf16.h>
#include <math.h>

// ---------------------------------------------------------------------------
// Gated linear-RNN layer for MI455X (gfx1250, wave32, WMMA bf16 datapath).
//   xn = rmsnorm(x) ; ab = xn @ W_in^T + b_in ; a = sigmoid, b = silu
//   h[n] = a[n]*h[n-1] + b[n] ; out = h @ W_out^T + b_out + x ; also h[-1]
// ---------------------------------------------------------------------------

typedef __attribute__((ext_vector_type(16))) __bf16 v16bf;
typedef __attribute__((ext_vector_type(2)))  __bf16 v2bf;
typedef __attribute__((ext_vector_type(8)))  float  v8f;

#define BS    4
#define SEQ   8192
#define DD    1024
#define DI    2048
#define MTOT  (BS * SEQ)   // 32768 rows
#define E2    (2 * DI)     // 4096

// Fast sigmoid: hardware v_exp_f32 + v_rcp_f32 (gate nonlinearity, rcp ulp fine).
__device__ __forceinline__ float sigmoidf(float v) {
    return __builtin_amdgcn_rcpf(1.0f + __expf(-v));
}

// Load one 16x32 bf16 fragment (A layout == B layout for row-major-along-K):
// lanes 0-15 hold row (row0+lane), K chunks [k0,k0+8) and [k0+16,k0+24);
// lanes 16-31 hold row (row0+lane-16), K chunks [k0+8,k0+16) and [k0+24,k0+32).
__device__ __forceinline__ v16bf load_frag(const __bf16* __restrict__ base,
                                           int row0, int lda, int k0, int lane) {
    int half = lane >> 4;
    int r    = row0 + (lane & 15);
    const __bf16* p = base + (size_t)r * lda + k0 + half * 8;
    v16bf f;
    ((uint4*)&f)[0] = *((const uint4*)p);        // K chunk 0 (8 bf16)
    ((uint4*)&f)[1] = *((const uint4*)(p + 16)); // K chunk 1 (8 bf16, +16 elems)
    return f;
}

// Wave computes a 32x64 tile: acc[2][4] accumulators, K stepped by 32.
__device__ __forceinline__ void mma_tile(const __bf16* __restrict__ A,
                                         const __bf16* __restrict__ B,
                                         int K, int m0, int n0, int lane,
                                         v8f acc[2][4]) {
    for (int k0 = 0; k0 < K; k0 += 32) {
        if (k0 + 32 < K) {  // WGP-scope global_prefetch_b8 on upcoming K chunks
            __builtin_prefetch(A + (size_t)(m0 + (lane & 15)) * K + k0 + 32, 0, 3);
            __builtin_prefetch(B + (size_t)(n0 + (lane & 15)) * K + k0 + 32, 0, 3);
        }
        v16bf aF0 = load_frag(A, m0,      K, k0, lane);
        v16bf aF1 = load_frag(A, m0 + 16, K, k0, lane);
        v16bf bF[4];
#pragma unroll
        for (int j = 0; j < 4; ++j) bF[j] = load_frag(B, n0 + 16 * j, K, k0, lane);
#pragma unroll
        for (int j = 0; j < 4; ++j) {
            acc[0][j] = __builtin_amdgcn_wmma_f32_16x16x32_bf16(
                false, aF0, false, bF[j], (short)0, acc[0][j], false, false);
            acc[1][j] = __builtin_amdgcn_wmma_f32_16x16x32_bf16(
                false, aF1, false, bF[j], (short)0, acc[1][j], false, false);
        }
    }
}

// ---------------------------------------------------------------------------
__global__ void __launch_bounds__(256) convert_kernel(const float* __restrict__ src,
                                                      __bf16* __restrict__ dst, int n) {
    int i = blockIdx.x * 256 + threadIdx.x;
    if (i < n) dst[i] = (__bf16)src[i];
}

// One block (8 waves) per row of x; LDS tree reduction for mean(x*x).
__global__ void __launch_bounds__(256) rmsnorm_kernel(const float* __restrict__ x,
                                                      const float* __restrict__ w,
                                                      __bf16* __restrict__ xn) {
    int m = blockIdx.x;
    const float* row = x + (size_t)m * DD;
    float ss = 0.0f;
    for (int i = threadIdx.x; i < DD; i += 256) {
        float v = row[i];
        ss += v * v;
    }
    __shared__ float red[256];
    red[threadIdx.x] = ss;
    __syncthreads();
    for (int s = 128; s > 0; s >>= 1) {
        if (threadIdx.x < s) red[threadIdx.x] += red[threadIdx.x + s];
        __syncthreads();
    }
    float rms = rsqrtf(red[0] * (1.0f / (float)DD) + 1e-6f);
    __bf16* orow = xn + (size_t)m * DD;
    for (int i = threadIdx.x; i < DD; i += 256) {
        orow[i] = (__bf16)(row[i] * rms * w[i]);
    }
}

// GEMM1: ab = xn(32768x1024) * W_in^T(1024x4096) + b_in ; gate + bf16 store.
__global__ void __launch_bounds__(256) gemm_in_kernel(const __bf16* __restrict__ xn,
                                                      const __bf16* __restrict__ w_in,
                                                      const float* __restrict__ b_in,
                                                      __bf16* __restrict__ aOut,
                                                      __bf16* __restrict__ bOut) {
    int lane = threadIdx.x & 31;
    int wave = threadIdx.x >> 5;
    int m0 = blockIdx.x * 128 + (wave & 3) * 32;
    int n0 = blockIdx.y * 128 + (wave >> 2) * 64;

    v8f acc[2][4] = {};
    mma_tile(xn, w_in, DD, m0, n0, lane, acc);

    int nIdx = lane & 15;
    int half = lane >> 4;
#pragma unroll
    for (int i = 0; i < 2; ++i) {
#pragma unroll
        for (int j = 0; j < 4; ++j) {
            int e = n0 + j * 16 + nIdx;   // column in [0, 4096)
            float bias = b_in[e];
#pragma unroll
            for (int v = 0; v < 8; ++v) {
                int m = m0 + i * 16 + v + half * 8;
                float val = acc[i][j][v] + bias;
                if (e < DI) {
                    aOut[(size_t)m * DI + e] = (__bf16)sigmoidf(val);                // sigmoid
                } else {
                    bOut[(size_t)m * DI + (e - DI)] = (__bf16)(val * sigmoidf(val)); // silu
                }
            }
        }
    }
}

// Sequential scan: one lane per PAIR of channels (packed bf16 dwords, 128B/txn),
// n unrolled x4 so 8 independent loads are in flight per iteration.
__global__ void __launch_bounds__(256) scan_kernel(const __bf16* __restrict__ a,
                                                   const __bf16* __restrict__ b,
                                                   const float* __restrict__ h_prev,
                                                   __bf16* __restrict__ hOut,
                                                   float* __restrict__ hLast) {
    int t  = blockIdx.x * 256 + threadIdx.x;     // 0 .. BS*DI/2 - 1
    int bi = t / (DI / 2);
    int ep = (t - bi * (DI / 2)) * 2;            // even channel index
    const unsigned* ap = (const unsigned*)(a + (size_t)bi * SEQ * DI + ep);
    const unsigned* bp = (const unsigned*)(b + (size_t)bi * SEQ * DI + ep);
    unsigned*       hp = (unsigned*)(hOut + (size_t)bi * SEQ * DI + ep);
    const int stride = DI / 2;                   // row stride in dwords

    float h0 = h_prev[bi * DI + ep];
    float h1 = h_prev[bi * DI + ep + 1];
    for (int n = 0; n < SEQ; n += 4) {
        unsigned av[4], bv[4];
#pragma unroll
        for (int u = 0; u < 4; ++u) {
            av[u] = ap[(size_t)(n + u) * stride];
            bv[u] = bp[(size_t)(n + u) * stride];
        }
#pragma unroll
        for (int u = 0; u < 4; ++u) {
            v2bf a2 = __builtin_bit_cast(v2bf, av[u]);
            v2bf b2 = __builtin_bit_cast(v2bf, bv[u]);
            h0 = fmaf((float)a2.x, h0, (float)b2.x);
            h1 = fmaf((float)a2.y, h1, (float)b2.y);
            v2bf h2 = { (__bf16)h0, (__bf16)h1 };
            hp[(size_t)(n + u) * stride] = __builtin_bit_cast(unsigned, h2);
        }
    }
    hLast[bi * DI + ep]     = h0;
    hLast[bi * DI + ep + 1] = h1;
}

// GEMM2: out = h(32768x2048) * W_out^T(2048x1024) + b_out + x (residual).
__global__ void __launch_bounds__(256) gemm_out_kernel(const __bf16* __restrict__ h,
                                                       const __bf16* __restrict__ w_out,
                                                       const float* __restrict__ b_out,
                                                       const float* __restrict__ x,
                                                       float* __restrict__ out) {
    int lane = threadIdx.x & 31;
    int wave = threadIdx.x >> 5;
    int m0 = blockIdx.x * 128 + (wave & 3) * 32;
    int n0 = blockIdx.y * 128 + (wave >> 2) * 64;

    v8f acc[2][4] = {};
    mma_tile(h, w_out, DI, m0, n0, lane, acc);

    int nIdx = lane & 15;
    int half = lane >> 4;
#pragma unroll
    for (int i = 0; i < 2; ++i) {
#pragma unroll
        for (int j = 0; j < 4; ++j) {
            int d = n0 + j * 16 + nIdx;   // column in [0, 1024)
            float bias = b_out[d];
#pragma unroll
            for (int v = 0; v < 8; ++v) {
                int m = m0 + i * 16 + v + half * 8;
                size_t idx = (size_t)m * DD + d;
                out[idx] = acc[i][j][v] + bias + x[idx];
            }
        }
    }
}

// ---------------------------------------------------------------------------
extern "C" void kernel_launch(void* const* d_in, const int* in_sizes, int n_in,
                              void* d_out, int out_size, void* d_ws, size_t ws_size,
                              hipStream_t stream) {
    const float* x      = (const float*)d_in[0];
    const float* h_prev = (const float*)d_in[1];
    const float* norm_w = (const float*)d_in[2];
    const float* W_in   = (const float*)d_in[3];
    const float* b_in   = (const float*)d_in[4];
    const float* W_out  = (const float*)d_in[5];
    const float* b_out  = (const float*)d_in[6];

    float* out    = (float*)d_out;
    float* h_last = out + (size_t)MTOT * DD;   // second output, concatenated

    char* ws = (char*)d_ws;
    __bf16* xn   = (__bf16*)ws;  ws += (size_t)MTOT * DD * 2;  //  64 MB
    __bf16* wi   = (__bf16*)ws;  ws += (size_t)E2 * DD * 2;    //   8 MB (L2-resident)
    __bf16* wo   = (__bf16*)ws;  ws += (size_t)DD * DI * 2;    //   4 MB (L2-resident)
    __bf16* aBuf = (__bf16*)ws;  ws += (size_t)MTOT * DI * 2;  // 128 MB
    __bf16* bBuf = (__bf16*)ws;  ws += (size_t)MTOT * DI * 2;  // 128 MB
    __bf16* hBuf = (__bf16*)ws;                                // 128 MB

    convert_kernel<<<(E2 * DD + 255) / 256, 256, 0, stream>>>(W_in, wi, E2 * DD);
    convert_kernel<<<(DD * DI + 255) / 256, 256, 0, stream>>>(W_out, wo, DD * DI);

    rmsnorm_kernel<<<MTOT, 256, 0, stream>>>(x, norm_w, xn);

    gemm_in_kernel<<<dim3(MTOT / 128, E2 / 128), 256, 0, stream>>>(xn, wi, b_in, aBuf, bBuf);

    scan_kernel<<<(BS * DI / 2) / 256, 256, 0, stream>>>(aBuf, bBuf, h_prev, hBuf, h_last);

    gemm_out_kernel<<<dim3(MTOT / 128, DD / 128), 256, 0, stream>>>(hBuf, wo, b_out, x, out);
}
